// PolyRnnHead_66503273611508
// MI455X (gfx1250) — compile-verified, matrix operand order checked by reading, not run on hardware
//
#include <hip/hip_runtime.h>
#include <hip/hip_bf16.h>
#include <math.h>
#include <stdint.h>

// ---------------------------------------------------------------------------
// PolyRNN head for MI455X (gfx1250, wave32, WMMA).
// Activations live in a zero-padded, PIXEL-MAJOR / CHANNEL-CONTIGUOUS layout:
//   element (ci, off) -> half index  off*ciPad + ci   (per image block).
// With the WMMA 16-bit lane map (lane-half h owns K halves [8h,8h+8) and
// [16+8h,16+8h+8)), every A/B fragment is exactly two aligned 128-bit loads.
// Matrix math: v_wmma_f32_16x16x32_f16 (f16 A/B, f32 accumulate).
// LDS staging uses the CDNA5 async global->LDS path (ASYNCcnt).
// ---------------------------------------------------------------------------

typedef __attribute__((ext_vector_type(16))) _Float16    v16h;
typedef __attribute__((ext_vector_type(8)))  float       v8f;
typedef __attribute__((ext_vector_type(4)))  unsigned int u32x4;

union Frag16 { v16h v; _Float16 h[16]; uint32_t u[8]; u32x4 q[2]; };

#define MODE_ACCUM 1
#define MODE_RELU  2
#define MODE_PAD16 4

__device__ __forceinline__ float sigm(float x) { return 1.0f / (1.0f + __expf(-x)); }
__device__ __forceinline__ int poff(int p) { return (p / 7 + 1) * 9 + (p % 7) + 1; }

// ---------------------------------------------------------------------------
// Generic conv-as-GEMM kernel.
//   feat : [n] blocks of 81(off) x ciPad(ci) halves (pixel-major)
//   wk   : [9][coTotal][ciPad] f16 packed weights (ci contiguous)
//   out  : f32 gates [n][coTotal][49]  OR  f16 pixel-major padded features.
// Block = 128 threads (4 waves). Grid = (n, coTotal/64).
// Wave w computes a 16(co) x 64(pixel, 49 valid) tile via 4 WMMA accumulators.
// ---------------------------------------------------------------------------
__global__ __launch_bounds__(128)
void convgemm_wmma(const _Float16* __restrict__ feat, int ciPad,
                   const _Float16* __restrict__ wk, int coTotal,
                   const float* __restrict__ bias,
                   float* __restrict__ gatesOut,
                   _Float16* __restrict__ padOut, int coStrideOut,
                   int mode)
{
  extern __shared__ _Float16 lds[];
  const int n   = blockIdx.x;
  const int tid = threadIdx.x;

  // Async stage of this image's feature block straight into LDS (no VGPR
  // round-trip): GLOBAL_LOAD_ASYNC_TO_LDS_B128, tracked by ASYNCcnt.
  {
    const size_t base = (size_t)n * ciPad * 81;     // halves (16B aligned)
    const int nchunks = (ciPad * 81) >> 3;          // 16-byte chunks
    const uint32_t ldsbase = (uint32_t)(uintptr_t)lds;
    for (int i = tid; i < nchunks; i += 128) {
      const _Float16* gp = feat + base + (size_t)i * 8;
      const uint32_t  la = ldsbase + (uint32_t)i * 16;
      asm volatile("global_load_async_to_lds_b128 %0, %1, off"
                   :: "v"(la), "v"(gp) : "memory");
    }
    asm volatile("s_wait_asynccnt 0x0" ::: "memory");
  }
  __syncthreads();

  const uint32_t* lds32 = (const uint32_t*)lds;
  const int ciHalf = ciPad >> 1;                  // dwords per pixel row
  const int wave = tid >> 5;
  const int lane = tid & 31;
  const int half = lane >> 4;
  const int l16  = lane & 15;
  const int coBase = blockIdx.y * 64 + wave * 16;

  int tOff[4];                                     // hh*9 + ww per tile
#pragma unroll
  for (int t = 0; t < 4; ++t) {
    int p = t * 16 + l16; if (p > 48) p = 48;      // clamp keeps EXEC uniform
    tOff[t] = (p / 7) * 9 + (p % 7);
  }

  const v8f vzero = {0.f, 0.f, 0.f, 0.f, 0.f, 0.f, 0.f, 0.f};
  v8f acc[4] = {vzero, vzero, vzero, vzero};

  const int ciChunks = ciPad >> 5;
#pragma unroll 1
  for (int k = 0; k < 9; ++k) {
    const int kOff = (k / 3) * 9 + (k % 3);        // uniform pixel shift
    const size_t wkRow = ((size_t)k * coTotal + coBase) * ciPad;
    __builtin_prefetch(wk + wkRow + (size_t)l16 * ciPad, 0, 0);
#pragma unroll 1
    for (int cc = 0; cc < ciChunks; ++cc) {
      // A fragment: two aligned 128-bit global loads.
      Frag16 a;
      const _Float16* ap = wk + wkRow + (size_t)l16 * ciPad + cc * 32 + half * 8;
      a.q[0] = *(const u32x4*)(ap);
      a.q[1] = *(const u32x4*)(ap + 16);

      // 4 B fragments: two aligned 128-bit LDS loads each.
      Frag16 b[4];
#pragma unroll
      for (int t = 0; t < 4; ++t) {
        const uint32_t* bp = lds32 + (size_t)(tOff[t] + kOff) * ciHalf
                                   + cc * 16 + half * 4;
        b[t].q[0] = *(const u32x4*)(bp);
        b[t].q[1] = *(const u32x4*)(bp + 8);
      }
#pragma unroll
      for (int t = 0; t < 4; ++t)
        acc[t] = __builtin_amdgcn_wmma_f32_16x16x32_f16(
            false, a.v, false, b[t].v, (short)0, acc[t], false, false);
    }
  }

  // Epilogue (divergence allowed now; all WMMAs retired).
#pragma unroll
  for (int t = 0; t < 4; ++t) {
    const int p = t * 16 + l16;
    if (p >= 49) continue;
    const int po = poff(p);
#pragma unroll
    for (int r = 0; r < 8; ++r) {
      const int m  = r + half * 8;
      const int co = coBase + m;
      float val = acc[t][r];
      if (bias) val += bias[co];
      if (mode & MODE_PAD16) {
        if (mode & MODE_RELU) val = val > 0.f ? val : 0.f;
        const size_t o = (size_t)n * 81 * coStrideOut + (size_t)po * coStrideOut + co;
        padOut[o] = (_Float16)val;
      } else {
        const size_t o = ((size_t)n * coTotal + co) * 49 + p;
        if (mode & MODE_ACCUM) gatesOut[o] += val;
        else                   gatesOut[o]  = val;
      }
    }
  }
}

// ---------------------------------------------------------------------------
// FC: logits[n][50] = h1flat . W^T, K reordered pixel-major (k = p*64 + c)
// so both A and B fragments are two aligned 128-bit global loads.
// Block = 128 (4 waves); each wave owns a 64(co) x 16(image) tile.
// ---------------------------------------------------------------------------
__global__ __launch_bounds__(128)
void fcgemm_wmma(const _Float16* __restrict__ h1pad, // [n] x 81(off) x 64(ch)
                 const _Float16* __restrict__ fcw,   // [64][3136], K pixel-major
                 const float* __restrict__ fcb,
                 float* __restrict__ out,            // [n][9][50]
                 int tstep)
{
  const int tid  = threadIdx.x;
  const int wave = tid >> 5, lane = tid & 31, half = lane >> 4, l16 = lane & 15;
  const int n0   = (blockIdx.x * 4 + wave) * 16;

  const v8f vzero = {0.f, 0.f, 0.f, 0.f, 0.f, 0.f, 0.f, 0.f};
  v8f acc[4] = {vzero, vzero, vzero, vzero};

  const int img = n0 + l16;
  const uint32_t* bimg = (const uint32_t*)(h1pad + (size_t)img * 81 * 64);

#pragma unroll 1
  for (int cc = 0; cc < 98; ++cc) {       // 98 * 32 = 3136 = 49 px * 64 ch
    const int p  = cc >> 1;               // pixel for this chunk
    const int c0 = (cc & 1) * 32;         // channel base for this chunk
    const uint32_t* bp = bimg + poff(p) * 32 + (c0 >> 1) + half * 4;
    Frag16 b;
    b.q[0] = *(const u32x4*)(bp);
    b.q[1] = *(const u32x4*)(bp + 8);
#pragma unroll
    for (int mt = 0; mt < 4; ++mt) {
      Frag16 a;
      const _Float16* ap = fcw + (size_t)(mt * 16 + l16) * 3136 + cc * 32 + half * 8;
      a.q[0] = *(const u32x4*)(ap);
      a.q[1] = *(const u32x4*)(ap + 16);
      acc[mt] = __builtin_amdgcn_wmma_f32_16x16x32_f16(
          false, a.v, false, b.v, (short)0, acc[mt], false, false);
    }
  }

#pragma unroll
  for (int mt = 0; mt < 4; ++mt)
#pragma unroll
    for (int r = 0; r < 8; ++r) {
      const int co = mt * 16 + r + half * 8;
      if (co >= 50) continue;
      out[((size_t)img * 9 + tstep) * 50 + co] = acc[mt][r] + fcb[co];
    }
}

// ------------------------------ pointwise -----------------------------------

__global__ void lstm_update(const float* __restrict__ g, float* __restrict__ c,
                            _Float16* __restrict__ hpad, int total)
{
  const int i = blockIdx.x * blockDim.x + threadIdx.x;
  if (i >= total) return;
  const int p  = i % 49;
  const int t1 = i / 49;
  const int ch = t1 % 64;
  const int n  = t1 / 64;
  const size_t gb = (size_t)n * 256 * 49 + p;
  const float gi = g[gb + (size_t)(ch      ) * 49];
  const float gf = g[gb + (size_t)(ch +  64) * 49];
  const float gg = g[gb + (size_t)(ch + 128) * 49];
  const float go = g[gb + (size_t)(ch + 192) * 49];
  const float cy = sigm(gf) * c[i] + sigm(gi) * tanhf(gg);
  c[i] = cy;
  hpad[(size_t)n * 81 * 64 + (size_t)poff(p) * 64 + ch] =
      (_Float16)(sigm(go) * tanhf(cy));
}

__global__ void argmax_update(const float* __restrict__ out, int tstep,
                              int* __restrict__ pm1, int* __restrict__ pm2,
                              _Float16* __restrict__ inp, int N)
{
  const int n = blockIdx.x * blockDim.x + threadIdx.x;
  if (n >= N) return;
  const float* lg = out + ((size_t)n * 9 + tstep) * 50;
  int best = 0; float bv = lg[0];
  for (int k = 1; k < 50; ++k) { const float v = lg[k]; if (v > bv) { bv = v; best = k; } }
  const int p2 = pm1[n];
  pm1[n] = best; pm2[n] = p2;
  // Channels 128 (v_pred2) and 129 (v_pred1) are adjacent halves -> one dword.
  _Float16* blk = inp + (size_t)n * 81 * 160;
  uint32_t* vmap = (uint32_t*)blk;
  for (int p = 0; p < 49; ++p) vmap[poff(p) * 80 + 64] = 0u;  // clear both maps
  if (p2   < 49) blk[poff(p2)   * 160 + 128] = (_Float16)1.f; // v_pred2
  if (best < 49) blk[poff(best) * 160 + 129] = (_Float16)1.f; // v_pred1
}

__global__ void init_vertex(const int* __restrict__ first,
                            int* __restrict__ pm1, int* __restrict__ pm2,
                            _Float16* __restrict__ inp, int N)
{
  const int n = blockIdx.x * blockDim.x + threadIdx.x;
  if (n >= N) return;
  const int fv = first[n];
  pm1[n] = fv; pm2[n] = 49;  // sentinel 49 => all-zero grid
  if (fv < 49) {
    _Float16* blk = inp + (size_t)n * 81 * 160;
    const int po = poff(fv);
    blk[po * 160 + 129] = (_Float16)1.f;  // v_pred1 = v_first
    blk[po * 160 + 130] = (_Float16)1.f;  // v_first
  }
}

// ------------------------------- packers ------------------------------------

__global__ void zero_u32(uint32_t* p, size_t ndw) {
  size_t i = (size_t)blockIdx.x * blockDim.x + threadIdx.x;
  const size_t st = (size_t)gridDim.x * blockDim.x;
  for (; i < ndw; i += st) p[i] = 0u;
}

__global__ void pack_x(const float* __restrict__ x, _Float16* __restrict__ xp,
                       size_t total /* N*256*49 */)
{
  const size_t i = (size_t)blockIdx.x * blockDim.x + threadIdx.x;
  if (i >= total) return;
  const int p = (int)(i % 49);
  const size_t nc = i / 49;
  const int c = (int)(nc % 256);
  const size_t n = nc / 256;
  xp[n * 81 * 256 + (size_t)poff(p) * 256 + c] = (_Float16)x[i];
}

__global__ void pack_conv_w(const float* __restrict__ w, _Float16* __restrict__ wk,
                            int Co, int Ci, int CiPad)
{
  const size_t i = (size_t)blockIdx.x * blockDim.x + threadIdx.x;
  const size_t total = (size_t)9 * Co * CiPad;
  if (i >= total) return;
  const int ci = (int)(i % CiPad);
  const size_t t = i / CiPad;
  const int co = (int)(t % Co);
  const int k  = (int)(t / Co);
  const float v = (ci < Ci) ? w[((size_t)co * Ci + ci) * 9 + k] : 0.f;
  wk[i] = (_Float16)v;
}

__global__ void pack_fc(const float* __restrict__ w, _Float16* __restrict__ fw)
{
  const size_t i = (size_t)blockIdx.x * blockDim.x + threadIdx.x;
  if (i >= (size_t)64 * 3136) return;
  const int k  = (int)(i % 3136);     // pixel-major: k = p*64 + c
  const int co = (int)(i / 3136);
  const int c = k % 64, p = k / 64;
  fw[i] = (_Float16)(co < 50 ? w[(size_t)co * 3136 + c * 49 + p] : 0.f);
}

// ------------------------------- launcher -----------------------------------

extern "C" void kernel_launch(void* const* d_in, const int* in_sizes, int n_in,
                              void* d_out, int out_size, void* d_ws, size_t ws_size,
                              hipStream_t stream)
{
  (void)n_in; (void)out_size; (void)ws_size;
  const float* x        = (const float*)d_in[0];
  const int*   firstv   = (const int*)  d_in[1];
  const float* conv0_w  = (const float*)d_in[2];
  const float* conv0_b  = (const float*)d_in[3];
  const float* conv1_w  = (const float*)d_in[4];
  const float* conv1_b  = (const float*)d_in[5];
  const float* cx0_w    = (const float*)d_in[6];
  const float* cx0_b    = (const float*)d_in[7];
  const float* ch0_w    = (const float*)d_in[8];
  const float* ch0_b    = (const float*)d_in[9];
  const float* cx1_w    = (const float*)d_in[10];
  const float* cx1_b    = (const float*)d_in[11];
  const float* ch1_w    = (const float*)d_in[12];
  const float* ch1_b    = (const float*)d_in[13];
  const float* fc_w     = (const float*)d_in[14];
  const float* fc_b     = (const float*)d_in[15];
  float* out = (float*)d_out;

  const int N = in_sizes[1];          // 2048
  char* ws = (char*)d_ws;

  // ---- workspace layout (256B aligned regions) ----
  size_t off = 0;
  auto take = [&](size_t bytes) { size_t r = off; off = (off + bytes + 255) & ~(size_t)255; return r; };
  const size_t o_xpack = take((size_t)N * 256 * 81 * 2);  // conv0 input, f16 pixel-major
  const size_t o_f0    = take((size_t)N * 128 * 81 * 2);  // conv0 out
  const size_t o_inp   = take((size_t)N * 160 * 81 * 2);  // [conv1(128)|vp2|vp1|vfirst|pad..160]
  const size_t o_h0    = take((size_t)N *  64 * 81 * 2);
  const size_t o_h1    = take((size_t)N *  64 * 81 * 2);
  const size_t o_c0    = take((size_t)N *  64 * 49 * 4);
  const size_t o_c1    = take((size_t)N *  64 * 49 * 4);
  const size_t zero_end = off;                            // everything above starts zeroed
  const size_t o_gate  = take((size_t)N * 256 * 49 * 4);  // shared gates buffer (fully written)
  const size_t o_wk0   = take((size_t)9 * 128 * 256 * 2);
  const size_t o_wk1   = take((size_t)9 * 128 * 128 * 2);
  const size_t o_wcx0  = take((size_t)9 * 256 * 160 * 2);
  const size_t o_wch0  = take((size_t)9 * 256 *  64 * 2);
  const size_t o_wcx1  = take((size_t)9 * 256 *  64 * 2);
  const size_t o_wch1  = take((size_t)9 * 256 *  64 * 2);
  const size_t o_fcw   = take((size_t)64 * 3136 * 2);
  const size_t o_pm1   = take((size_t)N * 4);
  const size_t o_pm2   = take((size_t)N * 4);

  _Float16* xpack = (_Float16*)(ws + o_xpack);
  _Float16* f0    = (_Float16*)(ws + o_f0);
  _Float16* inp   = (_Float16*)(ws + o_inp);
  _Float16* h0    = (_Float16*)(ws + o_h0);
  _Float16* h1    = (_Float16*)(ws + o_h1);
  float*    c0    = (float*)   (ws + o_c0);
  float*    c1    = (float*)   (ws + o_c1);
  float*    gates = (float*)   (ws + o_gate);
  _Float16* wk0   = (_Float16*)(ws + o_wk0);
  _Float16* wk1   = (_Float16*)(ws + o_wk1);
  _Float16* wcx0  = (_Float16*)(ws + o_wcx0);
  _Float16* wch0  = (_Float16*)(ws + o_wch0);
  _Float16* wcx1  = (_Float16*)(ws + o_wcx1);
  _Float16* wch1  = (_Float16*)(ws + o_wch1);
  _Float16* fcw   = (_Float16*)(ws + o_fcw);
  int*      pm1   = (int*)     (ws + o_pm1);
  int*      pm2   = (int*)     (ws + o_pm2);

  // ---- deterministic init: zero padded-feature & state regions every call ----
  zero_u32<<<4096, 256, 0, stream>>>((uint32_t*)ws, zero_end / 4);

  // ---- pack inputs / weights to f16 ----
  {
    const size_t tot = (size_t)N * 256 * 49;
    pack_x<<<(unsigned)((tot + 255) / 256), 256, 0, stream>>>(x, xpack, tot);
  }
  auto packW = [&](const float* w, _Float16* dst, int Co, int Ci, int CiPad) {
    const size_t tot = (size_t)9 * Co * CiPad;
    pack_conv_w<<<(unsigned)((tot + 255) / 256), 256, 0, stream>>>(w, dst, Co, Ci, CiPad);
  };
  packW(conv0_w, wk0, 128, 256, 256);
  packW(conv1_w, wk1, 128, 128, 128);
  packW(cx0_w, wcx0, 256, 131, 160);
  packW(ch0_w, wch0, 256,  64,  64);
  packW(cx1_w, wcx1, 256,  64,  64);
  packW(ch1_w, wch1, 256,  64,  64);
  pack_fc<<<(unsigned)(((size_t)64 * 3136 + 255) / 256), 256, 0, stream>>>(fc_w, fcw);
  init_vertex<<<(N + 255) / 256, 256, 0, stream>>>(firstv, pm1, pm2, inp, N);

  // ---- feature convs ----
  convgemm_wmma<<<dim3(N, 2), 128, 256 * 81 * 2, stream>>>(
      xpack, 256, wk0, 128, conv0_b, nullptr, f0, 128, MODE_PAD16 | MODE_RELU);
  convgemm_wmma<<<dim3(N, 2), 128, 128 * 81 * 2, stream>>>(
      f0, 128, wk1, 128, conv1_b, nullptr, inp, 160, MODE_PAD16 | MODE_RELU);

  // ---- recurrence: 9 serial steps (stream-ordered launches) ----
  const int lstmTotal = N * 64 * 49;
  for (int t = 0; t < 9; ++t) {
    convgemm_wmma<<<dim3(N, 4), 128, 160 * 81 * 2, stream>>>(
        inp, 160, wcx0, 256, cx0_b, gates, nullptr, 0, 0);
    convgemm_wmma<<<dim3(N, 4), 128, 64 * 81 * 2, stream>>>(
        h0, 64, wch0, 256, ch0_b, gates, nullptr, 0, MODE_ACCUM);
    lstm_update<<<(lstmTotal + 255) / 256, 256, 0, stream>>>(gates, c0, h0, lstmTotal);

    convgemm_wmma<<<dim3(N, 4), 128, 64 * 81 * 2, stream>>>(
        h0, 64, wcx1, 256, cx1_b, gates, nullptr, 0, 0);
    convgemm_wmma<<<dim3(N, 4), 128, 64 * 81 * 2, stream>>>(
        h1, 64, wch1, 256, ch1_b, gates, nullptr, 0, MODE_ACCUM);
    lstm_update<<<(lstmTotal + 255) / 256, 256, 0, stream>>>(gates, c1, h1, lstmTotal);

    fcgemm_wmma<<<N / 64, 128, 0, stream>>>(h1, fcw, fc_b, out, t);

    if (t < 8)
      argmax_update<<<(N + 255) / 256, 256, 0, stream>>>(out, t, pm1, pm2, inp, N);
  }
}
